// SentenceCompression_81243601371904
// MI455X (gfx1250) — compile-verified, matrix operand order checked by reading, not run on hardware
//
#include <hip/hip_runtime.h>
#include <stdint.h>
#include <stddef.h>

// ---------------- problem constants ----------------
#define B_    128
#define S_    128
#define I_    300
#define IPAD  320          // I padded to multiple of 32 (zero-filled)
#define H_    1024
#define NG    4096         // 4*H
#define KTOT  1344         // H + IPAD
#define KCH   (H_ / 32)    // 32 k-chunks over h
#define KCX   (IPAD / 32)  // 10 k-chunks over x
#define NBLK  32           // persistent blocks in recurrence kernel
#define UNITS 32           // hidden units owned per block (NBLK*UNITS == H)

typedef __bf16 bf16;
typedef __attribute__((ext_vector_type(16))) __bf16 v16bf;
typedef __attribute__((ext_vector_type(8)))  __bf16 v8bf;
typedef __attribute__((ext_vector_type(8)))  float  v8f;

static __device__ __forceinline__ v16bf load_frag(const bf16* p) {
    // lane holds 16 bf16: 8 contiguous at p, 8 contiguous at p+16 (ISA 16-bit A/B layout)
    v8bf lo = *(const v8bf*)(p);
    v8bf hi = *(const v8bf*)(p + 16);
    return __builtin_shufflevector(lo, hi, 0,1,2,3,4,5,6,7,8,9,10,11,12,13,14,15);
}

static __device__ __forceinline__ float sigmoidf_(float x) {
    return 1.0f / (1.0f + __expf(-x));
}
static __device__ __forceinline__ float ftanhf_(float x) {
    // tanh(x) = 1 - 2/(exp(2x)+1); single v_exp_f32 TRANS op
    return 1.0f - 2.0f / (__expf(2.0f * x) + 1.0f);
}

// ---------------- prep kernels ----------------

__global__ __launch_bounds__(256) void prep_weights_kernel(
    const float* __restrict__ W_ih, const float* __restrict__ W_hh,
    const float* __restrict__ b_ih, const float* __restrict__ b_hh,
    bf16* __restrict__ Wcomb, float* __restrict__ bsum)
{
    size_t idx = (size_t)blockIdx.x * blockDim.x + threadIdx.x;
    if (idx >= (size_t)NG * KTOT) return;
    int n = (int)(idx / KTOT);
    int k = (int)(idx % KTOT);
    float v;
    if (k < H_) {
        v = W_hh[(size_t)n * H_ + k];
    } else {
        int kk = k - H_;
        v = (kk < I_) ? W_ih[(size_t)n * I_ + kk] : 0.0f;
    }
    Wcomb[idx] = (bf16)v;
    if (k == 0) bsum[n] = b_ih[n] + b_hh[n];
}

__global__ __launch_bounds__(256) void prep_embed_kernel(
    const int* __restrict__ x, const float* __restrict__ embed_W,
    bf16* __restrict__ xe)
{
    size_t idx = (size_t)blockIdx.x * blockDim.x + threadIdx.x;
    if (idx >= (size_t)S_ * B_ * IPAD) return;
    int kk = (int)(idx % IPAD);
    int r  = (int)(idx / IPAD);       // r = s*B + b
    int s  = r / B_;
    int b  = r % B_;
    int tok = x[(size_t)b * S_ + s];
    float v = (kk < I_) ? embed_W[(size_t)tok * I_ + kk] : 0.0f;
    xe[idx] = (bf16)v;
}

__global__ __launch_bounds__(256) void prep_zero_kernel(
    bf16* __restrict__ h0, unsigned int* __restrict__ bar)
{
    size_t idx = (size_t)blockIdx.x * blockDim.x + threadIdx.x;
    if (idx < (size_t)B_ * H_) h0[idx] = (bf16)0.0f;
    if (idx == 0) *bar = 0u;
}

// ---------------- persistent LSTM recurrence ----------------
// 32 blocks x 512 threads (16 waves). Block b owns hidden units [b*32, b*32+32).
// Per step: gates[128 x 128] = [h_{t-1} | x_t] @ Wcomb[cols]^T + bsum  (K = 1344)
// Wave w: M-tile = w>>1 (16 batch rows), unit-half = w&1 (16 units), computes the
// 4 gate tiles (i,f,g,o) for those units -> acc is only 4 x v8f (32 VGPRs).
__global__ __launch_bounds__(512) void lstm_persistent_kernel(
    const bf16* __restrict__ Wcomb, const float* __restrict__ bsum,
    const bf16* __restrict__ xe, const bf16* __restrict__ h0,
    bf16* __restrict__ hs, unsigned int* __restrict__ bar)
{
    const int lane  = threadIdx.x & 31;
    const int wv    = threadIdx.x >> 5;
    const int mtile = wv >> 1;                // 0..7
    const int uh    = wv & 1;                 // unit-half 0/1
    const int col   = lane & 15;
    const int hiLn  = lane >> 4;              // 0: lanes 0-15, 1: lanes 16-31
    const int kbase = hiLn * 8;               // K sub-offset per ISA 16-bit layout
    const int unitbase = blockIdx.x * UNITS;

    // gate tile g (i,f,g,o) -> global gate column; preload bias (loop-invariant)
    int   ncol[4];
    float bias[4];
#pragma unroll
    for (int g = 0; g < 4; ++g) {
        ncol[g] = g * H_ + unitbase + uh * 16 + col;
        bias[g] = bsum[ncol[g]];
    }

    // persistent cell state for (row = mtile*16 + r + 8*hiLn, unit = unitbase + uh*16 + col)
    float cst[8];
#pragma unroll
    for (int r = 0; r < 8; ++r) cst[r] = 0.0f;

    const int m = mtile * 16 + col;           // batch row this lane loads for A

    for (int step = 0; step < S_; ++step) {
        const bf16* hprev = (step == 0) ? h0 : (hs + (size_t)(step - 1) * B_ * H_);
        const bf16* xet   = xe + (size_t)step * B_ * IPAD;

        v8f acc[4];
#pragma unroll
        for (int g = 0; g < 4; ++g) {
            float bv = bias[g];
#pragma unroll
            for (int e = 0; e < 8; ++e) acc[g][e] = bv;
        }

        // ---- K-part 1: recurrent term over h_{t-1} (32 chunks) ----
        {
            const bf16* ap  = hprev + (size_t)m * H_ + kbase;
            const bf16* bp0 = Wcomb + kbase;
#pragma unroll 1
            for (int kc = 0; kc < KCH; ++kc) {
                v16bf afrag = load_frag(ap);
#pragma unroll
                for (int g = 0; g < 4; ++g) {
                    v16bf bfrag = load_frag(bp0 + (size_t)ncol[g] * KTOT + (size_t)kc * 32);
                    acc[g] = __builtin_amdgcn_wmma_f32_16x16x32_bf16(
                        false, afrag, false, bfrag, (short)0, acc[g], false, false);
                }
                ap += 32;
            }
        }
        // ---- K-part 2: input term over x_t (10 chunks) ----
        {
            const bf16* ap  = xet + (size_t)m * IPAD + kbase;
            const bf16* bp0 = Wcomb + H_ + kbase;   // W_ih portion of combined rows
#pragma unroll 1
            for (int kc = 0; kc < KCX; ++kc) {
                v16bf afrag = load_frag(ap);
#pragma unroll
                for (int g = 0; g < 4; ++g) {
                    v16bf bfrag = load_frag(bp0 + (size_t)ncol[g] * KTOT + (size_t)kc * 32);
                    acc[g] = __builtin_amdgcn_wmma_f32_16x16x32_bf16(
                        false, afrag, false, bfrag, (short)0, acc[g], false, false);
                }
                ap += 32;
            }
        }

        // ---- elementwise gates + cell update + store h ----
        bf16* hcur = hs + (size_t)step * B_ * H_;
        const int outcol = unitbase + uh * 16 + col;
#pragma unroll
        for (int r = 0; r < 8; ++r) {
            int row = mtile * 16 + r + hiLn * 8;
            float ig = acc[0][r];   // input gate
            float fg = acc[1][r];   // forget gate
            float gg = acc[2][r];   // cell gate
            float og = acc[3][r];   // output gate
            float cnew = sigmoidf_(fg) * cst[r] + sigmoidf_(ig) * ftanhf_(gg);
            cst[r] = cnew;
            float hv = sigmoidf_(og) * ftanhf_(cnew);
            hcur[(size_t)row * H_ + outcol] = (bf16)hv;
        }

        // ---- device-wide barrier: all h stores visible before next step ----
        __threadfence();
        __syncthreads();
        if (threadIdx.x == 0) {
            atomicAdd(bar, 1u);
            unsigned target = (unsigned)(NBLK * (step + 1));
            while (__hip_atomic_load(bar, __ATOMIC_RELAXED, __HIP_MEMORY_SCOPE_AGENT) < target) {
                __builtin_amdgcn_s_sleep(2);
            }
        }
        __syncthreads();
        __threadfence();
    }
}

// ---------------- output projection + length mask ----------------
__global__ __launch_bounds__(256) void outproj_kernel(
    const bf16* __restrict__ hs, const float* __restrict__ linW,
    const float* __restrict__ linb, const int* __restrict__ lengths,
    float* __restrict__ out)
{
    const int lane = threadIdx.x & 31;
    const int wv   = threadIdx.x >> 5;
    const int R    = blockIdx.x * 8 + wv;
    if (R >= B_ * S_) return;
    const int b = R / S_;
    const int s = R % S_;
    const int len = lengths[b];
    if (s >= len) {
        if (lane == 0) { out[2 * (size_t)R] = 1.0f; out[2 * (size_t)R + 1] = 0.0f; }
        return;
    }
    const bf16* hp = hs + (size_t)s * B_ * H_ + (size_t)b * H_ + (size_t)lane * 32;
    float d0 = 0.0f, d1 = 0.0f;
#pragma unroll
    for (int c = 0; c < 4; ++c) {
        v8bf hv = *(const v8bf*)(hp + c * 8);
#pragma unroll
        for (int e = 0; e < 8; ++e) {
            float h = (float)hv[e];
            int k = lane * 32 + c * 8 + e;
            d0 += h * linW[k];
            d1 += h * linW[H_ + k];
        }
    }
#pragma unroll
    for (int off = 16; off > 0; off >>= 1) {
        d0 += __shfl_xor(d0, off, 32);
        d1 += __shfl_xor(d1, off, 32);
    }
    if (lane == 0) {
        out[2 * (size_t)R]     = d0 + linb[0];
        out[2 * (size_t)R + 1] = d1 + linb[1];
    }
}

// ---------------- launch ----------------
static inline size_t align256(size_t x) { return (x + 255) & ~(size_t)255; }

extern "C" void kernel_launch(void* const* d_in, const int* in_sizes, int n_in,
                              void* d_out, int out_size, void* d_ws, size_t ws_size,
                              hipStream_t stream) {
    (void)in_sizes; (void)n_in; (void)out_size; (void)ws_size;

    const int*   x       = (const int*)  d_in[0];
    const int*   lengths = (const int*)  d_in[1];
    const float* embed_W = (const float*)d_in[2];
    const float* W_ih    = (const float*)d_in[3];
    const float* W_hh    = (const float*)d_in[4];
    const float* b_ih    = (const float*)d_in[5];
    const float* b_hh    = (const float*)d_in[6];
    const float* lin_W   = (const float*)d_in[7];
    const float* lin_b   = (const float*)d_in[8];
    float* out = (float*)d_out;

    // workspace carve-up
    char* w = (char*)d_ws;
    bf16* Wcomb = (bf16*)w;                 w += align256((size_t)NG * KTOT * sizeof(bf16));
    bf16* xe    = (bf16*)w;                 w += align256((size_t)S_ * B_ * IPAD * sizeof(bf16));
    float* bsum = (float*)w;                w += align256((size_t)NG * sizeof(float));
    bf16* h0    = (bf16*)w;                 w += align256((size_t)B_ * H_ * sizeof(bf16));
    bf16* hsbuf = (bf16*)w;                 w += align256((size_t)S_ * B_ * H_ * sizeof(bf16));
    unsigned int* bar = (unsigned int*)w;   w += 256;

    {
        size_t n = (size_t)NG * KTOT;
        prep_weights_kernel<<<(unsigned)((n + 255) / 256), 256, 0, stream>>>(
            W_ih, W_hh, b_ih, b_hh, Wcomb, bsum);
    }
    {
        size_t n = (size_t)S_ * B_ * IPAD;
        prep_embed_kernel<<<(unsigned)((n + 255) / 256), 256, 0, stream>>>(x, embed_W, xe);
    }
    {
        size_t n = (size_t)B_ * H_;
        prep_zero_kernel<<<(unsigned)((n + 255) / 256), 256, 0, stream>>>(h0, bar);
    }
    lstm_persistent_kernel<<<NBLK, 512, 0, stream>>>(Wcomb, bsum, xe, h0, hsbuf, bar);
    outproj_kernel<<<(B_ * S_) / 8, 256, 0, stream>>>(hsbuf, lin_W, lin_b, lengths, out);
}
